// GPT2Block_49134425867052
// MI455X (gfx1250) — compile-verified, matrix operand order checked by reading, not run on hardware
//
#include <hip/hip_runtime.h>
#include <hip/hip_bf16.h>
#include <cmath>

typedef __attribute__((ext_vector_type(16))) _Float16 v16h;
typedef __attribute__((ext_vector_type(8)))  _Float16 v8h;
typedef __attribute__((ext_vector_type(8)))  float    v8f;
typedef __attribute__((ext_vector_type(4)))  unsigned int v4u;
typedef __attribute__((ext_vector_type(8)))  int      v8i;
typedef __attribute__((ext_vector_type(4)))  int      v4i;

#define B_     2
#define S_     4096
#define D_     768
#define H_     12
#define HD_    64
#define INNER_ 3072
#define M_     (B_*S_)   // 8192 rows

// Load a 16x32 f16 A-style WMMA fragment (ISA 7.12.2): lane (l&15) = row,
// (l>>4) selects K-halves {0-7,16-23} vs {8-15,24-31}. Caller passes the
// per-lane pointer to the first 8 contiguous halves; second group is +16.
__device__ inline v16h ldfrag(const _Float16* p) {
    union { v16h v; v8h h[2]; } u;
    u.h[0] = *(const v8h*)(p);
    u.h[1] = *(const v8h*)(p + 16);
    return u.v;
}

__device__ inline v8f wmma16(v16h a, v16h b, v8f c) {
    return __builtin_amdgcn_wmma_f32_16x16x32_f16(
        /*neg_a=*/false, a, /*neg_b=*/false, b,
        /*c_mod=*/(short)0, c, /*reuse_a=*/false, /*reuse_b=*/false);
}

// ---- Tensor Data Mover: DMA one 128x32 f16 tile (row stride K elems) into
// LDS with hardware padding: 16-DWORD rows + 4-DWORD pad -> ld = 40 halfs.
// D# packing per CDNA5 ISA 8.3/8.4 (group0 128b, group1 256b; groups 2/3
// unused for 2-D tiles). 6-arg builtin form (clang-23 / therock headers).
__device__ inline void tdm_load_tile(const _Float16* gptr, unsigned lds_byte_off, int K) {
    unsigned long long ga = (unsigned long long)(size_t)gptr;
    v4u g0;
    g0.x = 0x1u;                                      // count=1, user descriptor
    g0.y = lds_byte_off;                              // LDS dest (bytes)
    g0.z = (unsigned)(ga & 0xFFFFFFFFu);              // global_addr[31:0]
    g0.w = (unsigned)((ga >> 32) & 0x01FFFFFFu)       // global_addr[56:32]
         | 0x80000000u;                               // type=2 ("image")
    v8i g1;
    g1[0] = (int)((1u << 16)      // data_size = 2 bytes
                | (1u << 20)      // pad_enable
                | (3u << 22)      // pad_interval: 16 DWORDs (one 64B row)
                | (3u << 25));    // pad_amount:   4 DWORDs (16B) -> ld=40 halfs
    g1[1] = (int)(((unsigned)K & 0xFFFFu) << 16);     // tensor_dim0 = K (lo16)
    g1[2] = (int)0xFFFF0000u;                         // dim0 hi16=0 | tensor_dim1 lo16=0xFFFF
    g1[3] = (int)(32u << 16);                         // dim1 hi16=0 | tile_dim0 = 32
    g1[4] = 128;                                      // tile_dim1 = 128 | tile_dim2 = 0
    g1[5] = K;                                        // tensor_dim0_stride lo32 = K
    g1[6] = 0;                                        // stride0 hi | stride1 lo
    g1[7] = 0;                                        // stride1 hi
    v4i z4 = {0, 0, 0, 0};
    v8i z8 = {0, 0, 0, 0, 0, 0, 0, 0};
    __builtin_amdgcn_tensor_load_to_lds(g0, g1, z4, z4, z8, 0);
}

// ---------------- fp32 weight [K][N] -> f16 transposed [N][K] ----------------
__global__ __launch_bounds__(256) void wt_cvt_kernel(const float* __restrict__ w,
                                                     _Float16* __restrict__ wt,
                                                     int K, int N) {
    int idx = blockIdx.x * 256 + threadIdx.x;
    if (idx >= K * N) return;
    int k = idx / N, n = idx - k * N;
    wt[(size_t)n * K + k] = (_Float16)w[idx];
}

// ---------------- LayerNorm over D=768, one row per block ----------------
__global__ __launch_bounds__(256) void ln_kernel(const float* __restrict__ x,
                                                 const float* __restrict__ g,
                                                 const float* __restrict__ bta,
                                                 _Float16* __restrict__ out) {
    __shared__ float r0[8], r1[8];
    int row = blockIdx.x;
    int tid = threadIdx.x;
    const float* xr = x + (size_t)row * D_;
    float v0 = xr[tid], v1 = xr[tid + 256], v2 = xr[tid + 512];
    float s = v0 + v1 + v2;
    float q = v0 * v0 + v1 * v1 + v2 * v2;
    #pragma unroll
    for (int off = 16; off >= 1; off >>= 1) {
        s += __shfl_down(s, off, 32);
        q += __shfl_down(q, off, 32);
    }
    if ((tid & 31) == 0) { r0[tid >> 5] = s; r1[tid >> 5] = q; }
    __syncthreads();
    float ts = 0.f, tq = 0.f;
    #pragma unroll
    for (int i = 0; i < 8; ++i) { ts += r0[i]; tq += r1[i]; }
    float mu  = ts * (1.0f / D_);
    float var = tq * (1.0f / D_) - mu * mu;
    float rs  = rsqrtf(var + 1e-5f);
    _Float16* orow = out + (size_t)row * D_;
    orow[tid]       = (_Float16)((v0 - mu) * rs * g[tid]       + bta[tid]);
    orow[tid + 256] = (_Float16)((v1 - mu) * rs * g[tid + 256] + bta[tid + 256]);
    orow[tid + 512] = (_Float16)((v2 - mu) * rs * g[tid + 512] + bta[tid + 512]);
}

// ---------------- WMMA GEMM: C[M,N] = A[M,K] * Wt[N,K]^T + bias ----------------
// 128x128 block tile, BK=32, 8 waves (4x2), wave tile 32x64 = 2x4 v8f accums.
// Global->LDS staging via TDM (tensor_load_to_lds), double-buffered, one
// barrier per k-step. Wave 0 issues DMA + s_wait_tensorcnt.
// mode 0: scatter QKV -> q[b,h,s,d], k[b,h,s,d], vT[b,h,d,s]  (f16)
// mode 1: outf = acc + bias + resid                            (f32)
// mode 2: outh = gelu_exact(acc + bias)                        (f16)
__global__ __launch_bounds__(256) void gemm_kernel(
        const _Float16* __restrict__ A, const _Float16* __restrict__ Wt,
        const float* __restrict__ bias, int K, int mode,
        _Float16* __restrict__ qb, _Float16* __restrict__ kb,
        _Float16* __restrict__ vtb,
        const float* __restrict__ resid, float* __restrict__ outf,
        _Float16* __restrict__ outh) {
    // layout (halfs): [buf][tile] tiles of 128*40; buf stride 10240 halfs
    __shared__ __align__(128) _Float16 ls[2 * 2 * 128 * 40];
    int tid  = threadIdx.x;
    int lane = tid & 31, wv = tid >> 5;
    int wm = wv & 3, wn = wv >> 2;            // 4 waves in M, 2 in N
    int ln = lane & 15, hl = lane >> 4;
    int bm = blockIdx.y, bn = blockIdx.x;

    const _Float16* Abase = A  + (size_t)(bm * 128) * K;   // 128 rows, stride K
    const _Float16* Bbase = Wt + (size_t)(bn * 128) * K;
    unsigned ldsBase = (unsigned)(size_t)(void*)ls;        // LDS byte offset
    const unsigned TILE_B = 128 * 40 * 2;                  // 10240 bytes
    int nk = K >> 5;

    if (wv == 0) {
        tdm_load_tile(Abase, ldsBase, K);
        tdm_load_tile(Bbase, ldsBase + TILE_B, K);
    }

    v8f acc[2][4] = {};
    for (int kt = 0; kt < nk; ++kt) {
        int buf = kt & 1;
        if (wv == 0) __builtin_amdgcn_s_wait_tensorcnt(0);
        __syncthreads();   // data ready; prev buffer's readers are done
        if (wv == 0 && kt + 1 < nk) {
            unsigned nb = ldsBase + (unsigned)((kt + 1) & 1) * 2 * TILE_B;
            tdm_load_tile(Abase + (kt + 1) * 32, nb, K);
            tdm_load_tile(Bbase + (kt + 1) * 32, nb + TILE_B, K);
        }
        const _Float16* la = ls + (size_t)buf * 2 * 128 * 40;
        const _Float16* lb = la + 128 * 40;
        v16h af[2], bf[4];
        #pragma unroll
        for (int mi = 0; mi < 2; ++mi)
            af[mi] = ldfrag(&la[(wm * 32 + mi * 16 + ln) * 40 + hl * 8]);
        #pragma unroll
        for (int ni = 0; ni < 4; ++ni)
            bf[ni] = ldfrag(&lb[(wn * 64 + ni * 16 + ln) * 40 + hl * 8]);
        #pragma unroll
        for (int mi = 0; mi < 2; ++mi)
            #pragma unroll
            for (int ni = 0; ni < 4; ++ni)
                acc[mi][ni] = wmma16(af[mi], bf[ni], acc[mi][ni]);
    }

    // Epilogue. C layout: lane l, vgpr j -> row = j + (l>>4)*8, col = l&15.
    #pragma unroll
    for (int mi = 0; mi < 2; ++mi) {
        int rbase = bm * 128 + wm * 32 + mi * 16 + hl * 8;
        #pragma unroll
        for (int ni = 0; ni < 4; ++ni) {
            int col = bn * 128 + wn * 64 + ni * 16 + ln;
            float bv = bias[col];
            #pragma unroll
            for (int j = 0; j < 8; ++j) {
                int r = rbase + j;
                float v = acc[mi][ni][j] + bv;
                if (mode == 0) {
                    int part = col / D_;
                    int rem  = col - part * D_;
                    int hh = rem >> 6, dd = rem & 63;
                    int bb = r >> 12, ss = r & (S_ - 1);
                    size_t bh = (size_t)(bb * H_ + hh);
                    if (part == 0)      qb [(bh * S_  + ss) * HD_ + dd] = (_Float16)v;
                    else if (part == 1) kb [(bh * S_  + ss) * HD_ + dd] = (_Float16)v;
                    else                vtb[(bh * HD_ + dd) * S_  + ss] = (_Float16)v;
                } else if (mode == 1) {
                    size_t o = (size_t)r * D_ + col;
                    outf[o] = v + resid[o];
                } else {
                    float gg = 0.5f * v * (1.0f + erff(v * 0.70710678118654752f));
                    outh[(size_t)r * INNER_ + col] = (_Float16)gg;
                }
            }
        }
    }
}

// ---------------- Fused flash attention (causal, scale=1/8) ----------------
// 8 waves/block, each wave owns a 16-query strip; streams 32-key chunks with
// online softmax. ctx written as f16 in [B,S,D] layout for c_proj.
__global__ __launch_bounds__(256) void attn_kernel(
        const _Float16* __restrict__ qb, const _Float16* __restrict__ kb,
        const _Float16* __restrict__ vtb, _Float16* __restrict__ ctx) {
    __shared__ _Float16 lsP[8 * 16 * 40];   // per-wave 16x32 P tile (ld=40)
    int tid  = threadIdx.x;
    int lane = tid & 31, wv = tid >> 5;
    int ln = lane & 15, hl = lane >> 4;
    int blk  = blockIdx.x;
    int bh   = blk >> 5;                    // (b*H + h), 24 total
    int qblk = blk & 31;                    // 128-query block within S
    int qbase = qblk * 128 + wv * 16;

    const _Float16* qp = qb  + ((size_t)bh * S_ + qbase) * HD_;
    const _Float16* kp = kb  + (size_t)bh * S_ * HD_;
    const _Float16* vp = vtb + (size_t)bh * HD_ * S_;
    _Float16* pP = &lsP[wv * 16 * 40];

    v16h aQ0 = ldfrag(qp + ln * HD_ + hl * 8);
    v16h aQ1 = ldfrag(qp + ln * HD_ + 32 + hl * 8);

    v8f cacc[4] = {};
    float mr[8], lr[8];
    #pragma unroll
    for (int j = 0; j < 8; ++j) { mr[j] = -INFINITY; lr[j] = 0.f; }

    int kend = qbase + 15;                  // last key needed by this strip
    for (int cb = 0; cb <= kend; cb += 32) {
        // scores: Q(16x64) @ K^T -> two 16x16 accums (keys cb..cb+31)
        v16h bk0a = ldfrag(kp + (size_t)(cb + ln) * HD_ + hl * 8);
        v16h bk0b = ldfrag(kp + (size_t)(cb + ln) * HD_ + 32 + hl * 8);
        v16h bk1a = ldfrag(kp + (size_t)(cb + 16 + ln) * HD_ + hl * 8);
        v16h bk1b = ldfrag(kp + (size_t)(cb + 16 + ln) * HD_ + 32 + hl * 8);
        v8f s0 = {}, s1 = {};
        s0 = wmma16(aQ0, bk0a, s0); s0 = wmma16(aQ1, bk0b, s0);
        s1 = wmma16(aQ0, bk1a, s1); s1 = wmma16(aQ1, bk1b, s1);

        float p0[8], p1[8];
        #pragma unroll
        for (int j = 0; j < 8; ++j) {
            int r  = qbase + hl * 8 + j;
            int k0 = cb + ln, k1 = cb + 16 + ln;
            p0[j] = (k0 <= r) ? s0[j] * 0.125f : -1e9f;
            p1[j] = (k1 <= r) ? s1[j] * 0.125f : -1e9f;
        }
        #pragma unroll
        for (int j = 0; j < 8; ++j) {
            float mv = fmaxf(p0[j], p1[j]);
            #pragma unroll
            for (int off = 8; off >= 1; off >>= 1)
                mv = fmaxf(mv, __shfl_xor(mv, off, 16));
            float mn = fmaxf(mr[j], mv);
            float sc = expf(mr[j] - mn);    // 0 on first chunk (mr=-inf)
            mr[j] = mn;
            p0[j] = expf(p0[j] - mn);
            p1[j] = expf(p1[j] - mn);
            float sv = p0[j] + p1[j];
            #pragma unroll
            for (int off = 8; off >= 1; off >>= 1)
                sv += __shfl_xor(sv, off, 16);
            lr[j] = lr[j] * sc + sv;
            #pragma unroll
            for (int ni = 0; ni < 4; ++ni) cacc[ni][j] *= sc;
            // stage P (C-layout -> LDS row-major) for A-fragment reload
            pP[(hl * 8 + j) * 40 + ln]      = (_Float16)p0[j];
            pP[(hl * 8 + j) * 40 + 16 + ln] = (_Float16)p1[j];
        }
        v16h aP = ldfrag(&pP[ln * 40 + hl * 8]);   // same-wave LDS, DS in-order
        #pragma unroll
        for (int ni = 0; ni < 4; ++ni) {
            v16h bV = ldfrag(vp + (size_t)(ni * 16 + ln) * S_ + cb + hl * 8);
            cacc[ni] = wmma16(aP, bV, cacc[ni]);
        }
    }

    int bb = bh / H_, hh = bh - bb * H_;
    #pragma unroll
    for (int ni = 0; ni < 4; ++ni) {
        #pragma unroll
        for (int j = 0; j < 8; ++j) {
            int r = qbase + hl * 8 + j;
            float v = cacc[ni][j] / lr[j];
            ctx[((size_t)bb * S_ + r) * D_ + hh * HD_ + ni * 16 + ln] = (_Float16)v;
        }
    }
}

extern "C" void kernel_launch(void* const* d_in, const int* in_sizes, int n_in,
                              void* d_out, int out_size, void* d_ws, size_t ws_size,
                              hipStream_t stream) {
    (void)in_sizes; (void)n_in; (void)out_size; (void)ws_size;
    const float* hs    = (const float*)d_in[0];
    /* d_in[1] attention_mask: causal -1e9 mask reproduced analytically */
    const float* ln1g  = (const float*)d_in[2];
    const float* ln1b  = (const float*)d_in[3];
    const float* wattn = (const float*)d_in[4];
    const float* battn = (const float*)d_in[5];
    const float* wcp   = (const float*)d_in[6];
    const float* bcp   = (const float*)d_in[7];
    const float* ln2g  = (const float*)d_in[8];
    const float* ln2b  = (const float*)d_in[9];
    const float* wfc   = (const float*)d_in[10];
    const float* bfc   = (const float*)d_in[11];
    const float* wfp   = (const float*)d_in[12];
    const float* bfp   = (const float*)d_in[13];
    float* out = (float*)d_out;

    char* ws = (char*)d_ws;
    size_t o = 0;
    auto take = [&](size_t n) { size_t r = o; o += n; return r; };
    _Float16* wA  = (_Float16*)(ws + take((size_t)3 * D_ * D_ * 2));      // [2304][768]
    _Float16* wC  = (_Float16*)(ws + take((size_t)D_ * D_ * 2));          // [768][768]
    _Float16* wF  = (_Float16*)(ws + take((size_t)INNER_ * D_ * 2));      // [3072][768]
    _Float16* wP  = (_Float16*)(ws + take((size_t)D_ * INNER_ * 2));      // [768][3072]
    _Float16* xln = (_Float16*)(ws + take((size_t)M_ * D_ * 2));          // ln1/ln2 out
    _Float16* qb  = (_Float16*)(ws + take((size_t)M_ * D_ * 2));
    _Float16* kb  = (_Float16*)(ws + take((size_t)M_ * D_ * 2));
    _Float16* vtb = (_Float16*)(ws + take((size_t)M_ * D_ * 2));
    _Float16* ctx = (_Float16*)(ws + take((size_t)M_ * D_ * 2));
    float*    hb  = (float*)   (ws + take((size_t)M_ * D_ * 4));
    _Float16* fca = qb;   // q+k+vT+ctx (4 x 12.58MB) == M_*INNER_*2 exactly

    dim3 blk(256);
    // fp32 -> f16 transposed weights (once per launch; deterministic)
    wt_cvt_kernel<<<(D_ * 3 * D_ + 255) / 256, blk, 0, stream>>>(wattn, wA, D_, 3 * D_);
    wt_cvt_kernel<<<(D_ * D_ + 255) / 256,     blk, 0, stream>>>(wcp,   wC, D_, D_);
    wt_cvt_kernel<<<(D_ * INNER_ + 255) / 256, blk, 0, stream>>>(wfc,   wF, D_, INNER_);
    wt_cvt_kernel<<<(INNER_ * D_ + 255) / 256, blk, 0, stream>>>(wfp,   wP, INNER_, D_);
    // LN1
    ln_kernel<<<M_, blk, 0, stream>>>(hs, ln1g, ln1b, xln);
    // QKV projection -> scattered q/k/vT
    gemm_kernel<<<dim3(3 * D_ / 128, M_ / 128), blk, 0, stream>>>(
        xln, wA, battn, D_, 0, qb, kb, vtb, nullptr, nullptr, nullptr);
    // fused causal flash attention
    attn_kernel<<<B_ * H_ * (S_ / 128), blk, 0, stream>>>(qb, kb, vtb, ctx);
    // c_proj + residual -> h (f32)
    gemm_kernel<<<dim3(D_ / 128, M_ / 128), blk, 0, stream>>>(
        ctx, wC, bcp, D_, 1, nullptr, nullptr, nullptr, hs, hb, nullptr);
    // LN2
    ln_kernel<<<M_, blk, 0, stream>>>(hb, ln2g, ln2b, xln);
    // FC + exact GELU
    gemm_kernel<<<dim3(INNER_ / 128, M_ / 128), blk, 0, stream>>>(
        xln, wF, bfc, D_, 2, nullptr, nullptr, nullptr, nullptr, nullptr, fca);
    // FC-proj + residual -> out (f32)
    gemm_kernel<<<dim3(D_ / 128, M_ / 128), blk, 0, stream>>>(
        fca, wP, bfp, INNER_, 1, nullptr, nullptr, nullptr, hb, out, nullptr);
}
